// ClassicalHopfield_71536975282663
// MI455X (gfx1250) — 8-bit matrix operand order repaired and checked by reading, compiled, not run on hardware
//
#include <hip/hip_runtime.h>


typedef __attribute__((ext_vector_type(8))) int v8i;

#define NN 8192
#define PP 64
#define NBLK (NN / 16)
#define THREADS 256

#define SWZ_SWAPX16 0x401f

__device__ __forceinline__ int sgn_pm1(float x) { return (x >= 0.f) ? 1 : -1; }

__device__ __forceinline__ int pack4(int b0, int b1, int b2, int b3) {
  return (int)((unsigned)(b0 & 255) | ((unsigned)(b1 & 255) << 8) |
               ((unsigned)(b2 & 255) << 16) | ((unsigned)(b3 & 255) << 24));
}

__global__ __launch_bounds__(THREADS) void ClassicalHopfield_kernel(
    const float* __restrict__ X,
    const float* __restrict__ state,
    const int* __restrict__ perm,
    float* __restrict__ out)
{
  (void)perm;
  __shared__ __align__(16) int s_lds[NN];
  __shared__ __align__(16) int h_lds[PP];

  const int tid = threadIdx.x;

  for (int idx = tid; idx < NN; idx += THREADS)
    s_lds[idx] = sgn_pm1(state[idx]);
  if (tid < PP) h_lds[tid] = 0;
  __syncthreads();

  {
    const int p = tid & 63;
    const int g = tid >> 6;
    const int j0 = g * (NN / 4);
    int local = 0;
    for (int j = j0; j < j0 + NN / 4; ++j) {
      const int sj = s_lds[j];
      local += (X[(size_t)j * PP + p] >= 0.f) ? sj : -sj;
    }
    atomicAdd(&h_lds[p], local);
  }
  __syncthreads();

  if (tid >= 32) return;

  const int lane = tid;
  const int m  = lane & 15;
  const int hi = lane >> 4;

  for (int b = 0; b < NBLK; ++b) {
    const int i0 = b * 16;
    const float* rowM = X + (size_t)(i0 + m) * PP;

    float4 xfa[8];
    int accA = 0;
    #pragma unroll
    for (int v = 0; v < 8; ++v) {
      const int K = ((v >> 1) * 16) + (hi * 8) + ((v & 1) * 4);
      xfa[v] = *(const float4*)(rowM + K);
      const int4 hv = *(const int4*)(&h_lds[K]);
      accA += ((xfa[v].x >= 0.f) ? hv.x : -hv.x)
            + ((xfa[v].y >= 0.f) ? hv.y : -hv.y)
            + ((xfa[v].z >= 0.f) ? hv.z : -hv.z)
            + ((xfa[v].w >= 0.f) ? hv.w : -hv.w);
    }

    const int a_full = accA + __builtin_amdgcn_ds_swizzle(accA, SWZ_SWAPX16);

    const int s_old = s_lds[i0 + m];
    int c = a_full - PP * s_old;

    const int want = (c >= 0) ? 1 : -1;
    const unsigned flips = __builtin_amdgcn_ballot_w32(want != s_old);
    if (flips == 0u) {
      if (lane < 16) out[i0 + m] = (float)s_old;
      continue;
    }


    v8i A;
    #pragma unroll
    for (int v = 0; v < 8; ++v) {
      A[v] = pack4(sgn_pm1(xfa[v].x), sgn_pm1(xfa[v].y),
                   sgn_pm1(xfa[v].z), sgn_pm1(xfa[v].w));
    }

    v8i B;
    #pragma unroll
    for (int v = 0; v < 8; ++v) {
      const int K = ((v >> 1) * 16) + (hi * 8) + ((v & 1) * 4);
      const float4 xf = *(const float4*)(rowM + K);
      B[v] = pack4(sgn_pm1(xf.x), sgn_pm1(xf.y), sgn_pm1(xf.z), sgn_pm1(xf.w));
    }

    const v8i cz = {0, 0, 0, 0, 0, 0, 0, 0};
    v8i D = __builtin_amdgcn_wmma_i32_16x16x64_iu8(
         true, A,  true, B, cz,
         false,  false);

    int Gcol[16];
    #pragma unroll
    for (int r = 0; r < 8; ++r) {
      const int mine  = D[r];
      const int other = __builtin_amdgcn_ds_swizzle(mine, SWZ_SWAPX16);
      Gcol[r]     = (hi == 0) ? mine  : other;
      Gcol[r + 8] = (hi == 0) ? other : mine;
    }

    int deltas[16];
    int mynew = s_old;
    #pragma unroll
    for (int i = 0; i < 16; ++i) {
      const int act  = __builtin_amdgcn_readlane(c, i);
      const int olds = __builtin_amdgcn_readlane(s_old, i);
      const int news = (act >= 0) ? 1 : -1;
      const int d    = news - olds;
      deltas[i] = d;
      c += Gcol[i] * d;
      if (m == i) mynew = news;
    }

    #pragma unroll
    for (int t = 0; t < 2; ++t) {
      const int p = lane + t * 32;
      int dh = 0;
      #pragma unroll
      for (int j = 0; j < 16; ++j) {
        const float xv = X[(size_t)(i0 + j) * PP + p];
        dh += (xv >= 0.f) ? deltas[j] : -deltas[j];
      }
      h_lds[p] += dh;
    }

    if (lane < 16) {
      s_lds[i0 + m] = mynew;
      out[i0 + m] = (float)mynew;
    }
  }
}

extern "C" void kernel_launch(void* const* d_in, const int* in_sizes, int n_in,
                              void* d_out, int out_size, void* d_ws, size_t ws_size,
                              hipStream_t stream) {
  (void)in_sizes; (void)n_in; (void)out_size; (void)d_ws; (void)ws_size;
  const float* X     = (const float*)d_in[0];
  const float* state = (const float*)d_in[1];
  const int*   perm  = (const int*)d_in[2];
  float* out = (float*)d_out;
  hipLaunchKernelGGL(ClassicalHopfield_kernel, dim3(1), dim3(THREADS), 0, stream,
                     X, state, perm, out);
}
